// GtTransformer_51144470560931
// MI455X (gfx1250) — compile-verified, matrix operand order checked by reading, not run on hardware
//
#include <hip/hip_runtime.h>

// ---------------------------------------------------------------------------
// CDNA5 (gfx1250) transformer block: bf16 WMMA GEMMs, TDM-staged attention,
// K-split + f32 global atomics for the skinny-M FFN/classifier GEMMs.
// Shapes: B=256, S=64, D=512, H=8, DK=DV=64, DFF=2048, IN=32768, OUT=50
// ---------------------------------------------------------------------------

typedef __attribute__((ext_vector_type(16))) __bf16 v16bf;
typedef __attribute__((ext_vector_type(8)))  float  v8f;
typedef __attribute__((ext_vector_type(4)))  unsigned int v4u;
typedef __attribute__((ext_vector_type(8)))  int    v8i;
typedef __attribute__((ext_vector_type(4)))  int    v4i;

#define EPSLN 1e-5f

#if defined(__has_builtin)
#  if __has_builtin(__builtin_amdgcn_tensor_load_to_lds) && \
      __has_builtin(__builtin_amdgcn_s_wait_tensorcnt)
#    define HAVE_TDM 1
#  endif
#endif
#ifndef HAVE_TDM
#  define HAVE_TDM 0
#endif

// K index of pair p (0..7) for 16-bit WMMA A/B fragments (ISA 7.12.2):
// lanes 0-15 cover K {0..7, 16..23}; lanes 16-31 cover K {8..15, 24..31}
__device__ __forceinline__ int kpair(int p, int kbase) {
  return ((p & 4) << 2) + kbase + 2 * (p & 3);
}

#if HAVE_TDM
// Issue one TDM 2D tile load (fp32 elements) global -> LDS.
// D# group0/group1 bitfields per CDNA5 ISA.
// This toolchain's builtin takes 6 args (g0, g1, g2, g3, g4, cpol).
__device__ __forceinline__ void tdm_load_2d_f32(const float* gsrc, void* lds_dst,
                                                unsigned rows, unsigned cols,
                                                unsigned row_stride_elems) {
  unsigned long long ga = (unsigned long long)(uintptr_t)gsrc;
  unsigned lds = (unsigned)(uintptr_t)lds_dst;  // low 32 bits = LDS byte addr
  v4u g0;
  g0[0] = 1u;                                    // count=1, user descriptor
  g0[1] = lds;                                   // lds_addr
  g0[2] = (unsigned)(ga & 0xffffffffu);          // global_addr[31:0]
  g0[3] = (unsigned)((ga >> 32) & 0x1ffffffu) | (2u << 30);  // [56:32], type=2
  v8i g1;
  g1[0] = (int)(2u << 16);                       // data_size=2 (4 bytes)
  g1[1] = (int)((cols & 0xffffu) << 16);         // tensor_dim0[15:0]
  g1[2] = (int)((cols >> 16) | ((rows & 0xffffu) << 16));   // dim0 hi | dim1 lo
  g1[3] = (int)((rows >> 16) | (cols << 16));    // dim1 hi | tile_dim0=cols
  g1[4] = (int)rows;                             // tile_dim1=rows, tile_dim2=0
  g1[5] = (int)row_stride_elems;                 // tensor_dim0_stride[31:0]
  g1[6] = 0;                                     // stride0 hi | stride1 lo
  g1[7] = 0;
  v4i z4 = {0, 0, 0, 0};
  v8i z8 = {0, 0, 0, 0, 0, 0, 0, 0};
  __builtin_amdgcn_tensor_load_to_lds(g0, g1, z4, z4, z8, 0);
}
#endif

// ---------------------------------------------------------------------------
// Generic GEMM: D[M,N] = act(A[M,K] @ B[K,N] + bias), fp32 in/out, bf16 WMMA.
// Block tile 128x64 (8 waves, each 32x32 = 2x2 wmma tiles), K step 32.
// gridDim.z > 1 => K-split partial sums accumulated with global f32 atomics
// (bias/act applied by a separate epilogue kernel).
// ---------------------------------------------------------------------------
__global__ __launch_bounds__(256)
void gemm_bf16_wmma(const float* __restrict__ A, int lda,
                    const float* __restrict__ Bm, int ldb,
                    const float* __restrict__ bias,
                    float* __restrict__ D, int ldd,
                    int M, int N, int K, int relu)
{
  __shared__ __bf16 As[128 * 32];   // [m][k]   8 KB
  __shared__ __bf16 Bs[64 * 32];    // [n][k]   4 KB

  const int tid   = threadIdx.x;
  const int lane  = tid & 31;
  const int w     = tid >> 5;
  const int wm    = w & 3;          // 4 waves along M
  const int wn    = w >> 2;         // 2 waves along N
  const int mlane = lane & 15;
  const int kbase = (lane >> 4) * 8;

  const int row0 = blockIdx.x * 128;
  const int col0 = blockIdx.y * 64;

  const int kchunk = K / gridDim.z;        // divisible by 32 in all launches
  const int kbeg = blockIdx.z * kchunk;
  const int kend = kbeg + kchunk;

  v8f acc[2][2] = {};

  for (int k0 = kbeg; k0 < kend; k0 += 32) {
    // ---- stage A tile 128x32: float4 loads, packed bf16 cvt, b64 LDS stores
#pragma unroll
    for (int i = 0; i < 4; ++i) {
      int c = i * 256 + tid;               // float4 chunk 0..1023
      int m = c >> 3;                      // 8 chunks per 32-float row
      int kk4 = (c & 7) << 2;
      const float4 f4 =
          *(const float4*)&A[(size_t)(row0 + m) * lda + (k0 + kk4)];
      __bf16* dst = &As[m * 32 + kk4];
      dst[0] = (__bf16)f4.x; dst[1] = (__bf16)f4.y;
      dst[2] = (__bf16)f4.z; dst[3] = (__bf16)f4.w;
    }
    // ---- stage B tile 32x64 transposed into Bs[n][k]
    if (col0 + 64 <= N) {                  // fast path, 16B-aligned rows
#pragma unroll
      for (int i = 0; i < 2; ++i) {
        int c = i * 256 + tid;             // chunk 0..511
        int kk2 = c >> 4;                  // k row 0..31
        int n4 = (c & 15) << 2;
        const float4 f4 =
            *(const float4*)&Bm[(size_t)(k0 + kk2) * ldb + (col0 + n4)];
        Bs[(n4 + 0) * 32 + kk2] = (__bf16)f4.x;
        Bs[(n4 + 1) * 32 + kk2] = (__bf16)f4.y;
        Bs[(n4 + 2) * 32 + kk2] = (__bf16)f4.z;
        Bs[(n4 + 3) * 32 + kk2] = (__bf16)f4.w;
      }
    } else {                               // ragged N (classifier N=50)
#pragma unroll
      for (int i = 0; i < 8; ++i) {
        int idx = i * 256 + tid;
        int n = idx & 63, kk2 = idx >> 6;
        float val = (col0 + n < N) ? Bm[(size_t)(k0 + kk2) * ldb + (col0 + n)]
                                   : 0.0f;
        Bs[n * 32 + kk2] = (__bf16)val;
      }
    }
    __syncthreads();

    if (k0 + 32 < kend) {  // hint next K slab (global_prefetch_b8)
      __builtin_prefetch(&A[(size_t)(row0 + (tid >> 1)) * lda + (k0 + 32)], 0, 1);
      __builtin_prefetch(&Bm[(size_t)(k0 + 32 + (tid >> 3)) * ldb + col0], 0, 1);
    }

    v16bf afrag[2], bfrag[2];
#pragma unroll
    for (int t = 0; t < 2; ++t) {
      const __bf16* ap = &As[(wm * 32 + t * 16 + mlane) * 32];
      const __bf16* bp = &Bs[(wn * 32 + t * 16 + mlane) * 32];
#pragma unroll
      for (int p = 0; p < 8; ++p) {
        int kb = kpair(p, kbase);
        afrag[t][2 * p]     = ap[kb];
        afrag[t][2 * p + 1] = ap[kb + 1];
        bfrag[t][2 * p]     = bp[kb];
        bfrag[t][2 * p + 1] = bp[kb + 1];
      }
    }
#pragma unroll
    for (int tm = 0; tm < 2; ++tm)
#pragma unroll
      for (int tn = 0; tn < 2; ++tn)
        acc[tm][tn] = __builtin_amdgcn_wmma_f32_16x16x32_bf16(
            false, afrag[tm], false, bfrag[tn], (short)0, acc[tm][tn],
            false, false);
    __syncthreads();
  }

  // ---- epilogue: C/D layout -> row = r + 8*(lane>>4), col = lane&15
  const bool split = (gridDim.z > 1);
#pragma unroll
  for (int tm = 0; tm < 2; ++tm) {
#pragma unroll
    for (int tn = 0; tn < 2; ++tn) {
      int n = col0 + wn * 32 + tn * 16 + mlane;
      if (n >= N) continue;
      float bv = (!split && bias) ? bias[n] : 0.0f;
#pragma unroll
      for (int r = 0; r < 8; ++r) {
        int m = row0 + wm * 32 + tm * 16 + r + 8 * (lane >> 4);
        if (split) {
          atomicAdd(&D[(size_t)m * ldd + n], acc[tm][tn][r]);
        } else {
          float val = acc[tm][tn][r] + bv;
          if (relu) val = fmaxf(val, 0.0f);
          D[(size_t)m * ldd + n] = val;
        }
      }
    }
  }
}

// ---------------------------------------------------------------------------
// Fused attention for one (batch, head): Q/K/V tiles staged to LDS by the
// Tensor Data Mover (TDM), then scores -> softmax -> ctx, all bf16 WMMA.
// q/k/v layout: [H][B*S][64].  ctx layout: [B*S][H*64].
// ---------------------------------------------------------------------------
__global__ __launch_bounds__(256)
void attn_wmma(const float* __restrict__ q, const float* __restrict__ k,
               const float* __restrict__ v, float* __restrict__ ctx)
{
  __shared__ float sQ[64 * 64];   // 16 KB each
  __shared__ float sK[64 * 64];
  __shared__ float sV[64 * 64];
  __shared__ float P[64 * 64];

  const int b = blockIdx.x >> 3;
  const int h = blockIdx.x & 7;
  const int tid   = threadIdx.x;
  const int lane  = tid & 31;
  const int w     = tid >> 5;
  const int mlane = lane & 15;
  const int kbase = (lane >> 4) * 8;

  const size_t hoff = (size_t)h * 16384 * 64 + (size_t)b * 64 * 64;
  const float* qh = q + hoff;
  const float* kh = k + hoff;
  const float* vh = v + hoff;

#if HAVE_TDM
  if (w == 0) {                       // one wave drives the DMA engine
    tdm_load_2d_f32(qh, sQ, 64, 64, 64);
    tdm_load_2d_f32(kh, sK, 64, 64, 64);
    tdm_load_2d_f32(vh, sV, 64, 64, 64);
    __builtin_amdgcn_s_wait_tensorcnt(0);
  }
#else
  for (int i = tid; i < 64 * 64; i += 256) {
    sQ[i] = qh[i]; sK[i] = kh[i]; sV[i] = vh[i];
  }
#endif
  __syncthreads();

  // ---- scores = (Q @ K^T) / 8 : 4x4 grid of 16x16 tiles, 2 per wave ----
#pragma unroll
  for (int e = 0; e < 2; ++e) {
    int t = w * 2 + e;
    int ti = t >> 2, tj = t & 3;
    v8f acc = {};
#pragma unroll
    for (int kk = 0; kk < 64; kk += 32) {
      v16bf af, bfv;
      const float* ap = &sQ[(ti * 16 + mlane) * 64 + kk];
      // K^T fragment == A-style read of row-major K (lane n = row of K)
      const float* bp = &sK[(tj * 16 + mlane) * 64 + kk];
#pragma unroll
      for (int p = 0; p < 8; ++p) {
        int kb = kpair(p, kbase);
        af[2 * p]      = (__bf16)ap[kb];
        af[2 * p + 1]  = (__bf16)ap[kb + 1];
        bfv[2 * p]     = (__bf16)bp[kb];
        bfv[2 * p + 1] = (__bf16)bp[kb + 1];
      }
      acc = __builtin_amdgcn_wmma_f32_16x16x32_bf16(
          false, af, false, bfv, (short)0, acc, false, false);
    }
#pragma unroll
    for (int r = 0; r < 8; ++r) {
      int s = ti * 16 + r + 8 * (lane >> 4);
      P[s * 64 + tj * 16 + mlane] = acc[r] * 0.125f;  // 1/sqrt(DK)
    }
  }
  __syncthreads();

  // ---- softmax over rows (64 rows, one thread per row) ----
  if (tid < 64) {
    float mx = -1e30f;
    for (int j = 0; j < 64; ++j) mx = fmaxf(mx, P[tid * 64 + j]);
    float sum = 0.f;
    for (int j = 0; j < 64; ++j) {
      float ex = __expf(P[tid * 64 + j] - mx);
      P[tid * 64 + j] = ex;
      sum += ex;
    }
    float inv = 1.0f / sum;
    for (int j = 0; j < 64; ++j) P[tid * 64 + j] *= inv;
  }
  __syncthreads();

  // ---- ctx = P @ V ----
#pragma unroll
  for (int e = 0; e < 2; ++e) {
    int t = w * 2 + e;
    int ti = t >> 2, tj = t & 3;
    v8f acc = {};
#pragma unroll
    for (int kk = 0; kk < 64; kk += 32) {
      v16bf af, bfv;
      const float* ap = &P[(ti * 16 + mlane) * 64 + kk];
      const float* bp = &sV[kk * 64 + tj * 16 + mlane];
#pragma unroll
      for (int p = 0; p < 8; ++p) {
        int kb = kpair(p, kbase);
        af[2 * p]      = (__bf16)ap[kb];
        af[2 * p + 1]  = (__bf16)ap[kb + 1];
        bfv[2 * p]     = (__bf16)bp[kb * 64];
        bfv[2 * p + 1] = (__bf16)bp[(kb + 1) * 64];
      }
      acc = __builtin_amdgcn_wmma_f32_16x16x32_bf16(
          false, af, false, bfv, (short)0, acc, false, false);
    }
#pragma unroll
    for (int r = 0; r < 8; ++r) {
      int s  = ti * 16 + r + 8 * (lane >> 4);
      int dv = tj * 16 + mlane;
      ctx[((size_t)b * 64 + s) * 512 + h * 64 + dv] = acc[r];
    }
  }
}

// ---------------------------------------------------------------------------
// LayerNorm with residual: out[row] = LN(x[row] + resid[row]) * g + b
// ---------------------------------------------------------------------------
__global__ __launch_bounds__(256)
void ln_kernel(const float* __restrict__ x, const float* __restrict__ resid,
               const float* __restrict__ g, const float* __restrict__ bta,
               float* __restrict__ out, int n)
{
  __shared__ float red[256];
  const int row = blockIdx.x, tid = threadIdx.x;
  const float* xr = x + (size_t)row * n;
  const float* rr = resid + (size_t)row * n;

  float s = 0.f;
  for (int i = tid; i < n; i += 256) s += xr[i] + rr[i];
  red[tid] = s; __syncthreads();
  for (int st = 128; st > 0; st >>= 1) {
    if (tid < st) red[tid] += red[tid + st];
    __syncthreads();
  }
  const float mu = red[0] / n;
  __syncthreads();

  float vs = 0.f;
  for (int i = tid; i < n; i += 256) {
    float d = xr[i] + rr[i] - mu;
    vs += d * d;
  }
  red[tid] = vs; __syncthreads();
  for (int st = 128; st > 0; st >>= 1) {
    if (tid < st) red[tid] += red[tid + st];
    __syncthreads();
  }
  const float inv = rsqrtf(red[0] / n + EPSLN);

  float* orow = out + (size_t)row * n;
  for (int i = tid; i < n; i += 256)
    orow[i] = (xr[i] + rr[i] - mu) * inv * g[i] + bta[i];
}

// ---------------------------------------------------------------------------
__global__ __launch_bounds__(256)
void fill0(float* __restrict__ p, long n) {
  long i = (long)blockIdx.x * 256 + threadIdx.x;
  long stride = (long)gridDim.x * 256;
  for (; i < n; i += stride) p[i] = 0.0f;
}

__global__ __launch_bounds__(256)
void bias_act(float* __restrict__ x, const float* __restrict__ bias,
              int ncols, long total, int relu) {
  long i = (long)blockIdx.x * 256 + threadIdx.x;
  long stride = (long)gridDim.x * 256;
  for (; i < total; i += stride) {
    float v = x[i] + bias[(int)(i % ncols)];
    if (relu) v = fmaxf(v, 0.0f);
    x[i] = v;
  }
}

// ---------------------------------------------------------------------------
extern "C" void kernel_launch(void* const* d_in, const int* in_sizes, int n_in,
                              void* d_out, int out_size, void* d_ws, size_t ws_size,
                              hipStream_t stream) {
  const float* X   = (const float*)d_in[0];
  const float* Wq  = (const float*)d_in[1];
  const float* bq  = (const float*)d_in[2];
  const float* Wk  = (const float*)d_in[3];
  const float* bk  = (const float*)d_in[4];
  const float* Wv  = (const float*)d_in[5];
  const float* bv  = (const float*)d_in[6];
  const float* Wo  = (const float*)d_in[7];
  const float* bo  = (const float*)d_in[8];
  const float* g1  = (const float*)d_in[9];
  const float* be1 = (const float*)d_in[10];
  const float* W1  = (const float*)d_in[11];
  const float* b1  = (const float*)d_in[12];
  const float* W2  = (const float*)d_in[13];
  const float* b2  = (const float*)d_in[14];
  const float* g2  = (const float*)d_in[15];
  const float* be2 = (const float*)d_in[16];
  const float* Wf  = (const float*)d_in[17];
  const float* bfc = (const float*)d_in[18];
  float* out = (float*)d_out;

  float* ws = (float*)d_ws;
  const size_t BS = 16384;  // B*S
  float* qb   = ws;                         // [8][BS][64]
  float* kb   = qb   + (size_t)8 * BS * 64;
  float* vb   = kb   + (size_t)8 * BS * 64;
  float* ctx  = vb   + (size_t)8 * BS * 64; // [BS][512]
  float* aout = ctx  + BS * 512;            // [BS][512]
  float* hs   = aout + BS * 512;            // [BS][512]
  float* hf   = hs   + BS * 512;            // [256][2048]
  float* ypre = hf   + (size_t)256 * 2048;  // [256][32768]
  float* yb   = ypre + (size_t)256 * 32768; // [256][32768]

  dim3 blk(256);

  // QKV projections: per head, M=16384, N=64, K=512
  for (int h = 0; h < 8; ++h) {
    gemm_bf16_wmma<<<dim3(128, 1, 1), blk, 0, stream>>>(
        X, 512, Wq + (size_t)h * 512 * 64, 64, bq + h * 64,
        qb + (size_t)h * BS * 64, 64, 16384, 64, 512, 0);
    gemm_bf16_wmma<<<dim3(128, 1, 1), blk, 0, stream>>>(
        X, 512, Wk + (size_t)h * 512 * 64, 64, bk + h * 64,
        kb + (size_t)h * BS * 64, 64, 16384, 64, 512, 0);
    gemm_bf16_wmma<<<dim3(128, 1, 1), blk, 0, stream>>>(
        X, 512, Wv + (size_t)h * 512 * 64, 64, bv + h * 64,
        vb + (size_t)h * BS * 64, 64, 16384, 64, 512, 0);
  }

  // fused attention per (b,h)
  attn_wmma<<<dim3(256 * 8), blk, 0, stream>>>(qb, kb, vb, ctx);

  // output projection: M=16384, N=512, K=512
  gemm_bf16_wmma<<<dim3(128, 8, 1), blk, 0, stream>>>(
      ctx, 512, Wo, 512, bo, aout, 512, 16384, 512, 512, 0);

  // hs = LN(X + attn_out) over D=512
  ln_kernel<<<dim3(16384), blk, 0, stream>>>(aout, X, g1, be1, hs, 512);

  // FFN1: M=256, N=2048, K=32768, K-split x8 + atomic accumulate, then relu
  fill0<<<dim3(512), blk, 0, stream>>>(hf, 256L * 2048);
  gemm_bf16_wmma<<<dim3(2, 32, 8), blk, 0, stream>>>(
      hs, 32768, W1, 2048, nullptr, hf, 2048, 256, 2048, 32768, 0);
  bias_act<<<dim3(512), blk, 0, stream>>>(hf, b1, 2048, 256L * 2048, 1);

  // FFN2: M=256, N=32768, K=2048
  gemm_bf16_wmma<<<dim3(2, 512, 1), blk, 0, stream>>>(
      hf, 2048, W2, 32768, b2, ypre, 32768, 256, 32768, 2048, 0);

  // y = LN(ffn2 + hs) over 32768
  ln_kernel<<<dim3(256), blk, 0, stream>>>(ypre, hs, g2, be2, yb, 32768);

  // classifier: M=256, N=50, K=32768, K-split x16 + atomics, then bias
  fill0<<<dim3(50), blk, 0, stream>>>(out, 256L * 50);
  gemm_bf16_wmma<<<dim3(2, 1, 16), blk, 0, stream>>>(
      yb, 32768, Wf, 50, nullptr, out, 50, 256, 50, 32768, 0);
  bias_act<<<dim3(13), blk, 0, stream>>>(out, bfc, 50, 256L * 50, 0);
}
